// HyperLSTM_22960895164833
// MI455X (gfx1250) — matrix operand order repaired and checked by reading, not compiled, not run on hardware
//
#include <hip/hip_runtime.h>
#include <hip/hip_bf16.h>

#define BB   32
#define TT   512
#define DD   512
#define HMAIN 1024
#define HYP  128
#define NGATE 4

typedef __attribute__((ext_vector_type(16))) __bf16 v16bf;
typedef __attribute__((ext_vector_type(8)))  __bf16 v8bf;
typedef __attribute__((ext_vector_type(8)))  float  v8f;

// ---------------- WMMA fragment helpers (contiguous k order: k = 16*hsel + s) ------

__device__ __forceinline__ v16bf frag_cat(v8bf lo, v8bf hi) {
  return __builtin_shufflevector(lo, hi, 0,1,2,3,4,5,6,7,8,9,10,11,12,13,14,15);
}

// A (16x32 bf16): lanes 0-15 hold K = k0..k0+15 of row M=lane,
// lanes 16-31 hold K = k0+16..k0+31. Two 16B contiguous loads.
__device__ __forceinline__ v16bf load_a_frag(const __hip_bfloat16* row, int k0, int hsel) {
  v8bf lo = *(const v8bf*)(const void*)(row + k0 + hsel * 16);
  v8bf hi = *(const v8bf*)(const void*)(row + k0 + hsel * 16 + 8);
  return frag_cat(lo, hi);
}

// B (32x16 bf16), weights pre-transposed so column n is a contiguous row of K:
// lanes 0-15 hold K = k0..k0+15, lanes 16-31 hold K = k0+16..k0+31 of column n.
__device__ __forceinline__ v16bf load_b_frag(const __hip_bfloat16* colrow, int k0, int hsel) {
  v8bf lo = *(const v8bf*)(const void*)(colrow + k0 + hsel * 16);
  v8bf hi = *(const v8bf*)(const void*)(colrow + k0 + hsel * 16 + 8);
  return frag_cat(lo, hi);
}

__device__ __forceinline__ v8f wmma_bf16(v16bf a, v16bf b, v8f c) {
  return __builtin_amdgcn_wmma_f32_16x16x32_bf16(false, a, false, b, (short)0, c, false, false);
}

__device__ __forceinline__ float sigmoidf_(float x) { return 1.0f / (1.0f + __expf(-x)); }

// ---------------- data prep kernels ----------------------------------------

// x [B,T,D] f32 -> xbt [T,B,D] bf16
__global__ void k_conv_x(const float* __restrict__ x, __hip_bfloat16* __restrict__ xbt, int n) {
  int idx = blockIdx.x * blockDim.x + threadIdx.x;
  if (idx >= n) return;
  int b = idx / (TT * DD);
  int r = idx % (TT * DD);
  int t = r / DD, d = r % DD;
  xbt[((size_t)t * BB + b) * DD + d] = __float2bfloat16(x[idx]);
}

// Wm [1536,4096] f32 -> wmt [4096][1536] bf16 (transposed, K contiguous)
__global__ void k_conv_wmt(const float* __restrict__ Wm, __hip_bfloat16* __restrict__ wmt, int n) {
  int idx = blockIdx.x * blockDim.x + threadIdx.x;
  if (idx >= n) return;
  int col = idx / 1536, k = idx % 1536;
  wmt[idx] = __float2bfloat16(Wm[(size_t)k * 4096 + col]);
}

// W*_h [128,128] -> whht[g][n][k] bf16
__global__ void k_conv_whh(const float* __restrict__ w0, const float* __restrict__ w1,
                           const float* __restrict__ w2, const float* __restrict__ w3,
                           __hip_bfloat16* __restrict__ whht, int n) {
  int idx = blockIdx.x * blockDim.x + threadIdx.x;
  if (idx >= n) return;
  int g = idx >> 14, r = idx & 16383, nn = r >> 7, k = r & 127;
  const float* src = (g == 0) ? w0 : (g == 1) ? w1 : (g == 2) ? w2 : w3;
  whht[idx] = __float2bfloat16(src[k * 128 + nn]);
}

// W*_x [512,128] -> whxt[col=g*128+n][k] bf16
__global__ void k_conv_whx(const float* __restrict__ w0, const float* __restrict__ w1,
                           const float* __restrict__ w2, const float* __restrict__ w3,
                           __hip_bfloat16* __restrict__ whxt, int n) {
  int idx = blockIdx.x * blockDim.x + threadIdx.x;
  if (idx >= n) return;
  int col = idx >> 9, k = idx & 511;
  int g = col >> 7, nn = col & 127;
  const float* src = (g == 0) ? w0 : (g == 1) ? w1 : (g == 2) ? w2 : w3;
  whxt[idx] = __float2bfloat16(src[k * 128 + nn]);
}

__global__ void k_zero32(unsigned* __restrict__ p, int nwords) {
  int idx = blockIdx.x * blockDim.x + threadIdx.x;
  if (idx < nwords) p[idx] = 0u;
}

// ---------------- big parallel GEMM: XH = x @ [Wi_x|Wg_x|Wf_x|Wo_x] + bias --
// A [16384,512] bf16, Bt [512 cols][512 K] bf16 -> XH [16384,512] f32
__global__ __launch_bounds__(256) void k_hyperx(
    const __hip_bfloat16* __restrict__ xbt, const __hip_bfloat16* __restrict__ whxt,
    const float* __restrict__ b_i, const float* __restrict__ b_g,
    const float* __restrict__ b_f, const float* __restrict__ b_o,
    float* __restrict__ XH) {
  const int lane = threadIdx.x & 31, hsel = lane >> 4, l15 = lane & 15;
  const int tile = blockIdx.x * 8 + (threadIdx.x >> 5);
  const int rt = tile >> 5, ct = tile & 31;
  const int m0 = rt * 16, n0 = ct * 16;
  const __hip_bfloat16* arow = xbt + (size_t)(m0 + l15) * DD;
  const __hip_bfloat16* brow = whxt + (size_t)(n0 + l15) * DD;
  v8f acc = {0.f,0.f,0.f,0.f,0.f,0.f,0.f,0.f};
  for (int k0 = 0; k0 < DD; k0 += 32)
    acc = wmma_bf16(load_a_frag(arow, k0, hsel), load_b_frag(brow, k0, hsel), acc);
  const int nn = n0 + l15;
  const int g = nn >> 7, nc = nn & 127;
  const float* bp = (g == 0) ? b_i : (g == 1) ? b_g : (g == 2) ? b_f : b_o;
  const float bv = bp[nc];
  for (int r = 0; r < 8; ++r)
    XH[(size_t)(m0 + r + 8 * hsel) * 512 + nn] = acc[r] + bv;
}

// ---------------- per-step hyper-LSTM cell (1 block, 8 waves) ---------------
__global__ __launch_bounds__(256) void k_hyper_step(
    __hip_bfloat16* __restrict__ hhb,          // [32][128] bf16, in/out
    float* __restrict__ ch,                    // [32][128] f32, in/out
    const float* __restrict__ XH,              // [T*B, 512]
    const __hip_bfloat16* __restrict__ whht,   // [4][128][128]
    const float* __restrict__ lni_g, const float* __restrict__ lni_b,
    const float* __restrict__ lng_g, const float* __restrict__ lng_b,
    const float* __restrict__ lnf_g, const float* __restrict__ lnf_b,
    const float* __restrict__ lno_g, const float* __restrict__ lno_b,
    const float* __restrict__ lnc_g, const float* __restrict__ lnc_b,
    const float* __restrict__ pzx_w, const float* __restrict__ pzx_b,
    const float* __restrict__ pzh_w, const float* __restrict__ pzh_b,
    const float* __restrict__ pzb_w, const float* __restrict__ pzb_b,
    float* __restrict__ zall,                  // [3][32][16]
    int t) {
  __shared__ float act[NGATE][BB][HYP];        // 64 KB
  const int tid = threadIdx.x;
  const int lane = tid & 31, hsel = lane >> 4, l15 = lane & 15;
  const int w = tid >> 5;
  // phase 1: pre-activations = hh @ W*_h + XH[t]   (WMMA, 2 Mtiles x 8 Ntiles x 4 gates)
  {
    const int g = w >> 1, mt = w & 1;
    const __hip_bfloat16* arow = hhb + (size_t)(mt * 16 + l15) * HYP;
    for (int nt = 0; nt < 8; ++nt) {
      const __hip_bfloat16* brow = whht + ((size_t)g * HYP + nt * 16 + l15) * HYP;
      v8f acc = {0.f,0.f,0.f,0.f,0.f,0.f,0.f,0.f};
      for (int k0 = 0; k0 < HYP; k0 += 32)
        acc = wmma_bf16(load_a_frag(arow, k0, hsel), load_b_frag(brow, k0, hsel), acc);
      for (int r = 0; r < 8; ++r) {
        int m = mt * 16 + r + 8 * hsel;
        int n = nt * 16 + l15;
        act[g][m][n] = acc[r] + XH[((size_t)t * BB + m) * 512 + g * HYP + n];
      }
    }
  }
  __syncthreads();
  // phase 2: layernorm + activation per (gate,row)
  if (tid < 128) {
    int gg = tid >> 5, b = tid & 31;
    const float* lg = (gg == 0) ? lni_g : (gg == 1) ? lng_g : (gg == 2) ? lnf_g : lno_g;
    const float* lb = (gg == 0) ? lni_b : (gg == 1) ? lng_b : (gg == 2) ? lnf_b : lno_b;
    float s = 0.f, s2 = 0.f;
    for (int n = 0; n < HYP; ++n) { float v = act[gg][b][n]; s += v; s2 += v * v; }
    float mean = s * (1.0f / HYP);
    float var = s2 * (1.0f / HYP) - mean * mean;
    float inv = rsqrtf(var + 1e-5f);
    for (int n = 0; n < HYP; ++n) {
      float v = (act[gg][b][n] - mean) * inv * lg[n] + lb[n];
      act[gg][b][n] = (gg == 1) ? tanhf(v) : sigmoidf_(v);
    }
  }
  __syncthreads();
  // phase 3: ch_new = f*ch + i*g  (store into act[1])
  for (int idx = tid; idx < BB * HYP; idx += 256) {
    int m = idx >> 7, n = idx & 127;
    float cn = act[2][m][n] * ch[idx] + act[0][m][n] * act[1][m][n];
    ch[idx] = cn;
    act[1][m][n] = cn;
  }
  __syncthreads();
  // phase 4: hh_new = o * tanh(LN_c(ch_new))  (store into act[0], write bf16 state)
  if (tid < 32) {
    int b = tid;
    float s = 0.f, s2 = 0.f;
    for (int n = 0; n < HYP; ++n) { float v = act[1][b][n]; s += v; s2 += v * v; }
    float mean = s * (1.0f / HYP);
    float var = s2 * (1.0f / HYP) - mean * mean;
    float inv = rsqrtf(var + 1e-5f);
    for (int n = 0; n < HYP; ++n) {
      float v = (act[1][b][n] - mean) * inv * lnc_g[n] + lnc_b[n];
      float hv = act[3][b][n] * tanhf(v);
      act[0][b][n] = hv;
      hhb[b * HYP + n] = __float2bfloat16(hv);
    }
  }
  __syncthreads();
  // phase 5: z projections: zx/zh/zb = hh_new @ pz*_w + pz*_b   ([32,128]@[128,16])
  for (int idx = tid; idx < BB * 48; idx += 256) {
    int m = idx / 48, j = idx % 48;
    int sel = j >> 4, jj = j & 15;
    const float* wz = (sel == 0) ? pzx_w : (sel == 1) ? pzh_w : pzb_w;
    const float* bz = (sel == 0) ? pzx_b : (sel == 1) ? pzh_b : pzb_b;
    float s = bz[jj];
    for (int k = 0; k < HYP; ++k) s += act[0][m][k] * wz[k * 16 + jj];
    zall[sel * 512 + m * 16 + jj] = s;
  }
}

// ---------------- per-step main LSTM (32 blocks x 8 waves) ------------------
// Each block owns 32 h-columns; per wave: 2 chains, each = (gate, Mtile, Ntile),
// accX over K=512 (x part), accH over K=1024 (h part), fused hyper-scaling.
__global__ __launch_bounds__(256) void k_main_step(
    const __hip_bfloat16* __restrict__ xbt,    // [T*B, 512]
    const __hip_bfloat16* __restrict__ wmt,    // [4096][1536] transposed bf16
    const __hip_bfloat16* __restrict__ hb_cur, // [32][1024]
    __hip_bfloat16* __restrict__ hb_nxt,       // [32][1024]
    float* __restrict__ c,                     // [32][1024]
    const float* __restrict__ zall,            // [3][32][16]
    const float* __restrict__ pdx_w, const float* __restrict__ pdh_w,
    const float* __restrict__ pb_w,  const float* __restrict__ pb_b,
    float* __restrict__ out,                   // [B, T, H]
    int t) {
  __shared__ float gbuf[NGATE][BB][32];        // 16 KB
  __shared__ float zs[3 * 512];                // 6 KB
  const int tid = threadIdx.x;
  for (int i = tid; i < 3 * 512; i += 256) zs[i] = zall[i];
  __syncthreads();
  const int lane = tid & 31, hsel = lane >> 4, l15 = lane & 15, w = tid >> 5;
  const int n0 = blockIdx.x * 32;
  for (int cc = 0; cc < 2; ++cc) {
    const int idx16 = w * 2 + cc;
    const int g = idx16 & 3, mt = (idx16 >> 2) & 1, nt = (idx16 >> 3) & 1;
    const int mrow = mt * 16 + l15;
    const int nl = nt * 16 + l15;
    const int col = g * HMAIN + n0 + nl;
    const __hip_bfloat16* brow = wmt + (size_t)col * 1536;
    const __hip_bfloat16* arx = xbt + ((size_t)t * BB + mrow) * DD;
    const __hip_bfloat16* arh = hb_cur + (size_t)mrow * HMAIN;
    v8f accX = {0.f,0.f,0.f,0.f,0.f,0.f,0.f,0.f};
    v8f accH = {0.f,0.f,0.f,0.f,0.f,0.f,0.f,0.f};
    for (int k0 = 0; k0 < DD; k0 += 32)
      accX = wmma_bf16(load_a_frag(arx, k0, hsel), load_b_frag(brow, k0, hsel), accX);
    for (int k0 = 0; k0 < HMAIN; k0 += 32)
      accH = wmma_bf16(load_a_frag(arh, k0, hsel), load_b_frag(brow + DD, k0, hsel), accH);
    const int n = n0 + nl;
    for (int r = 0; r < 8; ++r) {
      int m = mt * 16 + r + 8 * hsel;
      float dx = 0.f, dh = 0.f, bd = pb_b[n];
      for (int e = 0; e < 4; ++e) {
        dx += zs[0 * 512 + m * 16 + g * 4 + e] * pdx_w[e * HMAIN + n];
        dh += zs[1 * 512 + m * 16 + g * 4 + e] * pdh_w[e * HMAIN + n];
        bd += zs[2 * 512 + m * 16 + g * 4 + e] * pb_w[e * HMAIN + n];
      }
      gbuf[g][m][nl] = accX[r] * dx + accH[r] * dh + bd;
    }
  }
  __syncthreads();
  // gates order from split: i, f, o, g
  for (int idx = tid; idx < BB * 32; idx += 256) {
    int m = idx >> 5, nl = idx & 31, n = n0 + nl;
    float gi = gbuf[0][m][nl], gf = gbuf[1][m][nl];
    float go = gbuf[2][m][nl], gz = gbuf[3][m][nl];
    float co = c[(size_t)m * HMAIN + n];
    float cn = sigmoidf_(gf) * co + sigmoidf_(gi) * tanhf(gz);
    float hn = sigmoidf_(go) * tanhf(cn);
    c[(size_t)m * HMAIN + n] = cn;
    hb_nxt[(size_t)m * HMAIN + n] = __float2bfloat16(hn);
    out[((size_t)m * TT + t) * HMAIN + n] = hn;
  }
}

// ---------------- final h/c tail copy ---------------------------------------
__global__ void k_finalize(float* __restrict__ out, const float* __restrict__ c) {
  int idx = blockIdx.x * blockDim.x + threadIdx.x;
  if (idx >= 2 * BB * HMAIN) return;
  const size_t BTH = (size_t)BB * TT * HMAIN;
  if (idx < BB * HMAIN) {
    int b = idx / HMAIN, n = idx % HMAIN;
    out[BTH + idx] = out[((size_t)b * TT + (TT - 1)) * HMAIN + n];
  } else {
    int j = idx - BB * HMAIN;
    out[BTH + (size_t)BB * HMAIN + j] = c[j];
  }
}

// ---------------- host launch ----------------------------------------------
extern "C" void kernel_launch(void* const* d_in, const int* in_sizes, int n_in,
                              void* d_out, int out_size, void* d_ws, size_t ws_size,
                              hipStream_t stream) {
  (void)in_sizes; (void)n_in; (void)out_size; (void)ws_size;
  const float* x     = (const float*)d_in[0];
  const float* Wm    = (const float*)d_in[1];
  const float* Wi_h  = (const float*)d_in[2];
  const float* Wi_x  = (const float*)d_in[3];
  const float* bi    = (const float*)d_in[4];
  const float* Wg_h  = (const float*)d_in[5];
  const float* Wg_x  = (const float*)d_in[6];
  const float* bg    = (const float*)d_in[7];
  const float* Wf_h  = (const float*)d_in[8];
  const float* Wf_x  = (const float*)d_in[9];
  const float* bf_   = (const float*)d_in[10];
  const float* Wo_h  = (const float*)d_in[11];
  const float* Wo_x  = (const float*)d_in[12];
  const float* bo    = (const float*)d_in[13];
  const float* lni_g = (const float*)d_in[14];
  const float* lni_b = (const float*)d_in[15];
  const float* lng_g = (const float*)d_in[16];
  const float* lng_b = (const float*)d_in[17];
  const float* lnf_g = (const float*)d_in[18];
  const float* lnf_b = (const float*)d_in[19];
  const float* lno_g = (const float*)d_in[20];
  const float* lno_b = (const float*)d_in[21];
  const float* lnc_g = (const float*)d_in[22];
  const float* lnc_b = (const float*)d_in[23];
  const float* pzx_w = (const float*)d_in[24];
  const float* pzx_b = (const float*)d_in[25];
  const float* pzh_w = (const float*)d_in[26];
  const float* pzh_b = (const float*)d_in[27];
  const float* pzb_w = (const float*)d_in[28];
  const float* pzb_b = (const float*)d_in[29];
  const float* pdx_w = (const float*)d_in[30];
  const float* pdh_w = (const float*)d_in[31];
  const float* pb_w  = (const float*)d_in[32];
  const float* pb_b  = (const float*)d_in[33];

  char* ws = (char*)d_ws;
  size_t off = 0;
  auto alloc = [&](size_t bytes) -> void* {
    void* p = (void*)(ws + off);
    off += (bytes + 255) & ~(size_t)255;
    return p;
  };
  __hip_bfloat16* xbt  = (__hip_bfloat16*)alloc((size_t)TT * BB * DD * 2);        // 16.8 MB
  __hip_bfloat16* wmt  = (__hip_bfloat16*)alloc((size_t)4096 * 1536 * 2);         // 12.6 MB
  __hip_bfloat16* whht = (__hip_bfloat16*)alloc((size_t)4 * 128 * 128 * 2);
  __hip_bfloat16* whxt = (__hip_bfloat16*)alloc((size_t)512 * 512 * 2);
  float*          XH   = (float*)alloc((size_t)TT * BB * 512 * 4);                // 33.6 MB
  char* state0 = ws + off;
  __hip_bfloat16* hb0  = (__hip_bfloat16*)alloc((size_t)BB * HMAIN * 2);
  __hip_bfloat16* hb1  = (__hip_bfloat16*)alloc((size_t)BB * HMAIN * 2);
  __hip_bfloat16* hhb  = (__hip_bfloat16*)alloc((size_t)BB * HYP * 2);
  float*          cbuf = (float*)alloc((size_t)BB * HMAIN * 4);
  float*          chb  = (float*)alloc((size_t)BB * HYP * 4);
  float*          zall = (float*)alloc((size_t)3 * BB * 16 * 4);
  size_t stateBytes = (size_t)((ws + off) - state0);

  int n;
  n = BB * TT * DD;
  k_conv_x<<<(n + 255) / 256, 256, 0, stream>>>(x, xbt, n);
  n = 4096 * 1536;
  k_conv_wmt<<<(n + 255) / 256, 256, 0, stream>>>(Wm, wmt, n);
  n = 4 * 128 * 128;
  k_conv_whh<<<(n + 255) / 256, 256, 0, stream>>>(Wi_h, Wg_h, Wf_h, Wo_h, whht, n);
  n = 512 * 512;
  k_conv_whx<<<(n + 255) / 256, 256, 0, stream>>>(Wi_x, Wg_x, Wf_x, Wo_x, whxt, n);
  n = (int)(stateBytes / 4);
  k_zero32<<<(n + 255) / 256, 256, 0, stream>>>((unsigned*)state0, n);

  // one big parallel WMMA GEMM: all x-dependent hyper-gate inputs
  k_hyperx<<<4096, 256, 0, stream>>>(xbt, whxt, bi, bg, bf_, bo, XH);

  for (int t = 0; t < TT; ++t) {
    k_hyper_step<<<1, 256, 0, stream>>>(
        hhb, chb, XH, whht,
        lni_g, lni_b, lng_g, lng_b, lnf_g, lnf_b, lno_g, lno_b, lnc_g, lnc_b,
        pzx_w, pzx_b, pzh_w, pzh_b, pzb_w, pzb_b, zall, t);
    __hip_bfloat16* cur = (t & 1) ? hb1 : hb0;
    __hip_bfloat16* nxt = (t & 1) ? hb0 : hb1;
    k_main_step<<<32, 256, 0, stream>>>(
        xbt, wmt, cur, nxt, cbuf, zall, pdx_w, pdh_w, pb_w, pb_b,
        (float*)d_out, t);
  }

  n = 2 * BB * HMAIN;
  k_finalize<<<(n + 255) / 256, 256, 0, stream>>>((float*)d_out, cbuf);
}
